// WindowMSA_35201551958425
// MI455X (gfx1250) — matrix operand order repaired and checked by reading, not compiled, not run on hardware
//
#include <hip/hip_runtime.h>
#include <hip/hip_bf16.h>

typedef _Float16 v16h __attribute__((ext_vector_type(16)));
typedef _Float16 v8h  __attribute__((ext_vector_type(8)));
typedef float    v8f  __attribute__((ext_vector_type(8)));
typedef unsigned int u32x4 __attribute__((ext_vector_type(4)));
typedef int          i32x8 __attribute__((ext_vector_type(8)));
typedef int          i32x4 __attribute__((ext_vector_type(4)));

#define NWIN   49      // tokens per window (7x7)
#define NPAD   64      // padded token count (4 tiles of 16)
#define CDIM   128     // embed dims
#define NHEAD  4
#define HD     32      // head dim
#define NMASK  64      // windows sharing one mask
#define QKV3   384     // 3*C
constexpr float ATTN_SCALE = 0.17677669529663687f; // 32^-0.5

// workspace layout (bytes)
#define WQKV_OFF  0                         // 384*128 f16  = 98304 B
#define WPROJ_OFF 98304                     // 128*128 f16  = 32768 B
#define BIAS_OFF  131072                    // 64*4*64*64 f32 = 4194304 B
#define WS_NEEDED (BIAS_OFF + (size_t)NMASK*NHEAD*NPAD*NPAD*4)

// LDS layout (bytes): sX/sO @0 (16K), sQ @16K, sK @32K (sP overlays sQ+sK),
// sVT @48K (feature-major 128x64), sS @64K (4 * 64 * 65 f32 = 66560)
#define SMEM_BYTES (65536 + 4*64*65*4)      // 132096

// ---- fragment helpers (ISA 7.12.2 layouts, wave32) -------------------------
__device__ inline v16h frag_join(const _Float16* p0, const _Float16* p1) {
    v8h lo = *(const v8h*)p0;
    v8h hi = *(const v8h*)p1;
    v16h r;
#pragma unroll
    for (int i = 0; i < 8; ++i) { r[i] = lo[i]; r[i + 8] = hi[i]; }
    return r;
}
// A 16x32 f16 (rows = M): element e -> K = 16*(e/8) + 8*hf + e%8
__device__ inline v16h loadA(const _Float16* base, int row, int stride,
                             int col0, int hf) {
    const _Float16* p = base + row * stride + col0 + 8 * hf;
    return frag_join(p, p + 16);
}
// B 32x16 f16 where B[k][n] = Src[n][k] (Src row-major over N): K = 16*(e/8) + 8*hf + e%8
__device__ inline v16h loadBT(const _Float16* base, int n0, int stride,
                              int k0, int lm, int hf) {
    const _Float16* p = base + (n0 + lm) * stride + k0 + 8 * hf;
    return frag_join(p, p + 16);
}

// ---- Tensor Data Mover: one descriptor copies the whole 4x64x64 f32 bias
// tile into LDS with hardware row padding (64 DW rows -> stride 65 DW).
// D# fields per CDNA5 ISA §8.3-8.6.
__device__ inline void tdm_load_bias(unsigned lds_addr, const float* gsrc) {
    unsigned long long ga = (unsigned long long)(size_t)gsrc;
    u32x4 g0;
    g0[0] = 1u;                                       // count=1, user mode
    g0[1] = lds_addr;                                 // lds_addr[31:0]
    g0[2] = (unsigned)ga;                             // global_addr[31:0]
    g0[3] = (unsigned)((ga >> 32) & 0x01FFFFFFu)      // global_addr[56:32]
            | (2u << 30);                             // type=2 ("image")
    i32x8 g1;
    g1[0] = (int)((2u << 16)      // data_size = 4 bytes
                | (1u << 20)      // pad_enable
                | (5u << 22));    // pad_interval: 64 DWORDs (pad_amount=0 -> 1 DW)
    g1[1] = (int)(64u << 16);     // tensor_dim0 = 64 (low16 in bits 63:48)
    g1[2] = (int)(256u << 16);    // tensor_dim1 = 256 (low16 in bits 95:80)
    g1[3] = (int)(64u << 16);     // tile_dim0 = 64 (bits 127:112)
    g1[4] = 256;                  // tile_dim1 = 256 (bits 143:128)
    g1[5] = 64;                   // tensor_dim0_stride = 64 (bits 191:160)
    g1[6] = (int)(16384u << 16);  // tensor_dim1_stride = 16384 (bits 223:208)
    g1[7] = 0;
    i32x4 gz = {0, 0, 0, 0};
#if __has_include(<hip/amd_detail/amd_gfx1250_TDM.h>)
    i32x8 gz8 = {0, 0, 0, 0, 0, 0, 0, 0};
    __builtin_amdgcn_tensor_load_to_lds(g0, g1, gz, gz, gz8, 0);
#else
    __builtin_amdgcn_tensor_load_to_lds(g0, g1, gz, gz, 0);
#endif
}

// ---- prep: f16 weights + folded bias table ---------------------------------
__global__ void wmsa_prep_kernel(const float* __restrict__ mask,
                                 const float* __restrict__ qkv_w,
                                 const float* __restrict__ rpb,
                                 const float* __restrict__ proj_w,
                                 char* __restrict__ ws) {
    _Float16* wq = (_Float16*)(ws + WQKV_OFF);
    _Float16* wp = (_Float16*)(ws + WPROJ_OFF);
    float* bias  = (float*)(ws + BIAS_OFF);
    const int NQ = QKV3 * CDIM, NP = CDIM * CDIM;
    const int NB = NMASK * NHEAD * NPAD * NPAD;
    const int total = NQ + NP + NB;
    for (int i = blockIdx.x * blockDim.x + threadIdx.x; i < total;
         i += gridDim.x * blockDim.x) {
        if (i < NQ) {
            wq[i] = (_Float16)qkv_w[i];
        } else if (i < NQ + NP) {
            wp[i - NQ] = (_Float16)proj_w[i - NQ];
        } else {
            int t = i - NQ - NP;
            int m = t & 63, n = (t >> 6) & 63, h = (t >> 12) & 3, w = t >> 14;
            float v = -1e30f;
            if (n < NWIN && m < NWIN) {
                int i1 = n / 7, j1 = n % 7, i2 = m / 7, j2 = m % 7;
                int ridx = (i1 - i2 + 6) * 13 + (j1 - j2 + 6);
                v = rpb[ridx * NHEAD + h] + mask[(w * NWIN + n) * NWIN + m];
            }
            bias[((w * NHEAD + h) * NPAD + n) * NPAD + m] = v;
        }
    }
}

// ---- fused window-MSA: 1 block = 1 window, 8 wave32s -----------------------
__global__ __launch_bounds__(256)
void wmsa_main_kernel(const float* __restrict__ x,
                      const float* __restrict__ qkv_b,
                      const float* __restrict__ proj_b,
                      const char* __restrict__ ws,
                      float* __restrict__ out) {
    extern __shared__ char smem[];
    _Float16* sX  = (_Float16*)smem;            // 64x128 token-major (-> sO)
    _Float16* sQ  = (_Float16*)(smem + 16384);  // 64x128 token-major, q*scale
    _Float16* sK  = (_Float16*)(smem + 32768);  // 64x128 token-major
    _Float16* sVT = (_Float16*)(smem + 49152);  // 128x64 feature-major
    float*    sS  = (float*)(smem + 65536);     // 4 x 64 x 65 f32
    _Float16* sP  = sQ;                         // 4 x 64 x 64 f16 (over sQ+sK)
    _Float16* sO  = sX;

    const _Float16* wq = (const _Float16*)(ws + WQKV_OFF);
    const _Float16* wp = (const _Float16*)(ws + WPROJ_OFF);
    const float* bias  = (const float*)(ws + BIAS_OFF);

    const int b    = blockIdx.x;
    const int tid  = threadIdx.x;
    const int wave = tid >> 5, lane = tid & 31;
    const int lm   = lane & 15, hf = lane >> 4;
    const float* xb    = x + (size_t)b * NWIN * CDIM;
    const float* biasb = bias + (size_t)(b & (NMASK - 1)) * NHEAD * NPAD * NPAD;

    // ---- Phase 0a: TDM-stage folded bias -> sS (HW-padded stride 65) ----
    // Single tensor_load_to_lds issued by wave 0; overlaps Phases 0b and 1.
    if (wave == 0)
        tdm_load_bias((unsigned)(size_t)sS, biasb);

    // ---- Phase 0b: x -> LDS f16 (rows >= 49 zeroed) ----
    for (int p = tid; p < NPAD * CDIM / 2; p += 256) {
        int row = p >> 6, c2 = (p & 63) * 2;
        _Float16 a0 = (_Float16)0.f, a1 = (_Float16)0.f;
        if (row < NWIN) {
            float2 v = *(const float2*)(xb + row * CDIM + c2);
            a0 = (_Float16)v.x; a1 = (_Float16)v.y;
        }
        sX[row * CDIM + c2]     = a0;
        sX[row * CDIM + c2 + 1] = a1;
    }
    __syncthreads();

    // ---- Phase 1: QKV^T = Wqkv @ X^T (+b). A = Wqkv rows (global, contig),
    //      B = X^T via token-major sX (contig). D: M=feature => packed stores.
    {
        v8f acc[3][4];
        const int j0 = wave * 3;   // feature tiles {3w..3w+2} x 4 token tiles
#pragma unroll
        for (int jj = 0; jj < 3; ++jj) {
            int f0 = (j0 + jj) * 16 + 8 * hf;
            float4 b0 = *(const float4*)(qkv_b + f0);
            float4 b1 = *(const float4*)(qkv_b + f0 + 4);
#pragma unroll
            for (int r = 0; r < 4; ++r) {
                acc[jj][r][0] = b0.x; acc[jj][r][1] = b0.y;
                acc[jj][r][2] = b0.z; acc[jj][r][3] = b0.w;
                acc[jj][r][4] = b1.x; acc[jj][r][5] = b1.y;
                acc[jj][r][6] = b1.z; acc[jj][r][7] = b1.w;
            }
        }
#pragma unroll
        for (int kc = 0; kc < 4; ++kc) {
            v16h Bf[4];
#pragma unroll
            for (int r = 0; r < 4; ++r)
                Bf[r] = loadBT(sX, r * 16, CDIM, kc * 32, lm, hf);
#pragma unroll
            for (int jj = 0; jj < 3; ++jj) {
                v16h Af = loadA(wq, (j0 + jj) * 16 + lm, CDIM, kc * 32, hf);
#pragma unroll
                for (int r = 0; r < 4; ++r)
                    acc[jj][r] = __builtin_amdgcn_wmma_f32_16x16x32_f16(
                        false, Af, false, Bf[r], (short)0, acc[jj][r],
                        false, false);
            }
        }
#pragma unroll
        for (int jj = 0; jj < 3; ++jj) {
            int fc = (j0 + jj) * 16;            // feature base in [0,384)
            if (fc < 256) {                     // q or k: token-major, packed
                _Float16* dst = (fc < 128) ? (sQ + fc) : (sK + fc - 128);
                float scl = (fc < 128) ? ATTN_SCALE : 1.0f;
#pragma unroll
                for (int r = 0; r < 4; ++r) {
                    v8h pk;
#pragma unroll
                    for (int e = 0; e < 8; ++e)
                        pk[e] = (_Float16)(acc[jj][r][e] * scl);
                    *(v8h*)(dst + (r * 16 + lm) * CDIM + 8 * hf) = pk;
                }
            } else {                            // v: feature-major sVT
                int d0 = fc - 256 + 8 * hf;
#pragma unroll
                for (int r = 0; r < 4; ++r)
#pragma unroll
                    for (int e = 0; e < 8; ++e)
                        sVT[(d0 + e) * NPAD + r * 16 + lm] =
                            (_Float16)acc[jj][r][e];
            }
        }
    }
    if (wave == 0)
        __builtin_amdgcn_s_wait_tensorcnt(0);   // bias tile landed in LDS
    __syncthreads();                            // visible to all waves

    // ---- Phase 2: S = q @ k^T + bias (C-init from LDS, in-place) ----
    {
        const int h = wave >> 1, rbase = (wave & 1) * 2;
        float* sSh = sS + h * (64 * 65);
        v16h Bf[4];
#pragma unroll
        for (int c = 0; c < 4; ++c)
            Bf[c] = loadBT(sK, c * 16, CDIM, h * HD, lm, hf);
#pragma unroll
        for (int rr = 0; rr < 2; ++rr) {
            int r = rbase + rr;
            v16h Af = loadA(sQ, r * 16 + lm, CDIM, h * HD, hf);
#pragma unroll
            for (int c = 0; c < 4; ++c) {
                v8f acc;
#pragma unroll
                for (int e = 0; e < 8; ++e)
                    acc[e] = sSh[(r * 16 + e + 8 * hf) * 65 + c * 16 + lm];
                acc = __builtin_amdgcn_wmma_f32_16x16x32_f16(
                    false, Af, false, Bf[c], (short)0, acc, false, false);
#pragma unroll
                for (int e = 0; e < 8; ++e)
                    sSh[(r * 16 + e + 8 * hf) * 65 + c * 16 + lm] = acc[e];
            }
        }
    }
    __syncthreads();

    // ---- Phase 3: row softmax (f32), write P f16 token-major ----
    {
        const int h = tid >> 6, n = tid & 63;
        const float* rowp = sS + h * (64 * 65) + n * 65;
        float mx = -3e38f;
        for (int m = 0; m < NPAD; ++m) mx = fmaxf(mx, rowp[m]);
        float sum = 0.f;
        for (int m = 0; m < NPAD; ++m) sum += __expf(rowp[m] - mx);
        float inv = 1.f / sum;
        _Float16* pr = sP + (h * 64 + n) * 64;
        for (int m = 0; m < NPAD; ++m)
            pr[m] = (_Float16)(__expf(rowp[m] - mx) * inv);
    }
    __syncthreads();

    // ---- Phase 4: O^T = V^T @ P^T per head. A = sVT rows (contig),
    //      B = P^T via token-major sP (contig). D packed into token-major sO.
    {
        const int h = wave >> 1, ft = wave & 1;
        const _Float16* Ph = sP + h * 64 * 64;
        const _Float16* Vh = sVT + (h * HD + ft * 16) * NPAD;
        v8f acc[4];
#pragma unroll
        for (int c = 0; c < 4; ++c)
#pragma unroll
            for (int e = 0; e < 8; ++e) acc[c][e] = 0.f;
#pragma unroll
        for (int kc = 0; kc < 2; ++kc) {
            v16h Af = loadA(Vh, lm, NPAD, kc * 32, hf);
#pragma unroll
            for (int c = 0; c < 4; ++c) {
                v16h Bf = loadBT(Ph, c * 16, 64, kc * 32, lm, hf);
                acc[c] = __builtin_amdgcn_wmma_f32_16x16x32_f16(
                    false, Af, false, Bf, (short)0, acc[c], false, false);
            }
        }
#pragma unroll
        for (int c = 0; c < 4; ++c) {
            v8h pk;
#pragma unroll
            for (int e = 0; e < 8; ++e) pk[e] = (_Float16)acc[c][e];
            *(v8h*)(sO + (c * 16 + lm) * CDIM + h * HD + ft * 16 + 8 * hf) = pk;
        }
    }
    __syncthreads();

    // ---- Phase 5: out^T = Wproj @ O^T (+b). A = Wproj rows (global, contig),
    //      B = O^T via token-major sO (contig). D: 8 consecutive features ->
    //      two global_store_b128 per tile, predicated on token < 49.
    {
        const int tt = wave >> 1, fb = (wave & 1) * 4;
        v8f acc[4];
#pragma unroll
        for (int cc = 0; cc < 4; ++cc) {
            int f0 = (fb + cc) * 16 + 8 * hf;
            float4 b0 = *(const float4*)(proj_b + f0);
            float4 b1 = *(const float4*)(proj_b + f0 + 4);
            acc[cc][0] = b0.x; acc[cc][1] = b0.y;
            acc[cc][2] = b0.z; acc[cc][3] = b0.w;
            acc[cc][4] = b1.x; acc[cc][5] = b1.y;
            acc[cc][6] = b1.z; acc[cc][7] = b1.w;
        }
#pragma unroll
        for (int kc = 0; kc < 4; ++kc) {
            v16h Bf = loadBT(sO, tt * 16, CDIM, kc * 32, lm, hf);
#pragma unroll
            for (int cc = 0; cc < 4; ++cc) {
                v16h Af = loadA(wp, (fb + cc) * 16 + lm, CDIM, kc * 32, hf);
                acc[cc] = __builtin_amdgcn_wmma_f32_16x16x32_f16(
                    false, Af, false, Bf, (short)0, acc[cc], false, false);
            }
        }
        int token = tt * 16 + lm;
        if (token < NWIN) {
            float* ob = out + ((size_t)b * NWIN + token) * CDIM;
#pragma unroll
            for (int cc = 0; cc < 4; ++cc) {
                int f0 = (fb + cc) * 16 + 8 * hf;
                float4 s0, s1;
                s0.x = acc[cc][0]; s0.y = acc[cc][1];
                s0.z = acc[cc][2]; s0.w = acc[cc][3];
                s1.x = acc[cc][4]; s1.y = acc[cc][5];
                s1.z = acc[cc][6]; s1.w = acc[cc][7];
                *(float4*)(ob + f0)     = s0;
                *(float4*)(ob + f0 + 4) = s1;
            }
        }
    }
}

extern "C" void kernel_launch(void* const* d_in, const int* in_sizes, int n_in,
                              void* d_out, int out_size, void* d_ws,
                              size_t ws_size, hipStream_t stream) {
    const float* x      = (const float*)d_in[0];
    const float* mask   = (const float*)d_in[1];
    const float* qkv_w  = (const float*)d_in[2];
    const float* qkv_b  = (const float*)d_in[3];
    const float* rpb    = (const float*)d_in[4];
    const float* proj_w = (const float*)d_in[5];
    const float* proj_b = (const float*)d_in[6];
    float* out = (float*)d_out;
    char* ws   = (char*)d_ws;
    if (ws_size < WS_NEEDED) return;

    (void)hipFuncSetAttribute((const void*)wmsa_main_kernel,
                              hipFuncAttributeMaxDynamicSharedMemorySize,
                              SMEM_BYTES);

    wmsa_prep_kernel<<<512, 256, 0, stream>>>(mask, qkv_w, rpb, proj_w, ws);

    const int nblocks = 4096;  // B windows
    wmsa_main_kernel<<<nblocks, 256, SMEM_BYTES, stream>>>(x, qkv_b, proj_b,
                                                           ws, out);
}
